// RNTN_77524159693658
// MI455X (gfx1250) — matrix operand order repaired and checked by reading, not compiled, not run on hardware
//
#include <hip/hip_runtime.h>
#include <hip/hip_bf16.h>
#include <math.h>

typedef __attribute__((ext_vector_type(16))) __bf16 v16bf;
typedef __attribute__((ext_vector_type(8)))  float  v8f;

#define B_DIM        1024
#define NODES_TOTAL  130048   // 1024*(64+32+16+8+4+2+1)
#define LEAF_ROWS    65536

#define TPB          128      // 4 waves
#define WAVES        4
#define M_PER_WAVE   32       // 2 accumulator sets of 16 nodes
#define NODES_PER_BLK (WAVES * M_PER_WAVE)   // 128

#define NCHUNK_V     512      // K-chunks of 32 over vec(c x d) = 16384
#define NCHUNK_W     4        // W-chunks (K over c = 128)
#define NCHUNK_PAD   520      // padded to 65 full stages of 8
#define CHUNK_USH    2048     // 4 ntiles * 32 lanes * 16 bf16
#define CHUNK_BYTES  4096
#define STAGE_CH     8        // chunks per LDS stage
#define NSTAGE_FULL  64       // stages 0..63 = Vt; stage 64 = W(4) + zero pad(4)
#define BBUF_BYTES   (STAGE_CH * CHUNK_BYTES)        // 32,768
#define XB_STRIDE    132      // bf16 row stride (bank-conflict padding)
#define XB_BYTES_PW  (M_PER_WAVE * XB_STRIDE * 2)    // 8,448
#define SMEM_LVL     (2 * BBUF_BYTES + WAVES * XB_BYTES_PW)   // 99,328 B

#define VT2_ELEMS    (NCHUNK_PAD * CHUNK_USH)        // 1,064,960 bf16

// ---------------------------------------------------------------------------
// Build the B operand stream: Bil = Outer(Mx16512) @ Bmat(16512x64).
//   chunks t<512   : (c = t>>2, dblk = t&3) -> B[K][k] = Vt[k][c][dblk*32+K]
//   512<=t<516     : ci = t-512             -> B[K][k] = W[ci*32+K][k]
//   516<=t<520     : zero padding (staged but never consumed)
// B fragment layout (32x16): lanes 0-15 -> N, K=0..15 (elem e<->K=e);
//                            lanes 16-31 -> K=16..31.
// Stream index: ((t*4 + ni)*32 + lane)*16 + e   (linear per-chunk blocks)
// ---------------------------------------------------------------------------
__global__ __launch_bounds__(256) void vt_swizzle_kernel(
    const float* __restrict__ Vt, const float* __restrict__ W,
    unsigned short* __restrict__ vt_sw)
{
  int idx  = blockIdx.x * 256 + threadIdx.x;   // exact grid
  int e    = idx & 15;
  int lane = (idx >> 4) & 31;
  int ni   = (idx >> 9) & 3;
  int t    = idx >> 11;
  int koff = e + ((lane >= 16) ? 16 : 0);      // B-layout K mapping
  int k    = ni * 16 + (lane & 15);
  float v;
  if (t < NCHUNK_V) {
    int c = t >> 2;
    int d = (t & 3) * 32 + koff;
    v = Vt[((size_t)k * 128 + c) * 128 + d];
  } else if (t < NCHUNK_V + NCHUNK_W) {
    int c = (t - NCHUNK_V) * 32 + koff;
    v = W[c * 64 + k];
  } else {
    v = 0.0f;                                  // pad chunks
  }
  unsigned int bits = __float_as_uint(v);      // f32 -> bf16 RNE
  unsigned int r    = bits + 0x7fffu + ((bits >> 16) & 1u);
  vt_sw[idx] = (unsigned short)(r >> 16);
}

// ---------------------------------------------------------------------------
__global__ __launch_bounds__(256) void gather_kernel(
    const int* __restrict__ leaf_ids, const float* __restrict__ embed,
    float* __restrict__ nodes)
{
  int idx = blockIdx.x * 256 + threadIdx.x;    // exact: 65536*64
  int row = idx >> 6;
  int c   = idx & 63;
  nodes[idx] = embed[(size_t)leaf_ids[row] * 64 + c];
}

// ---------------------------------------------------------------------------
// One tree level. 128 threads = 4 waves, 32 output nodes per wave (2 sets).
// Pure-WMMA bilinear+affine GEMM over K=16512 with on-the-fly outer-product
// A-fragments (v_pk_mul_bf16); B streamed via async DMA double buffer and
// software-pipelined by one chunk so LDS latency hides behind WMMA+VALU.
// ---------------------------------------------------------------------------
__global__ __launch_bounds__(TPB) void level_kernel(
    const float* __restrict__ nodes_in,
    float* __restrict__ nodes_out,
    const unsigned short* __restrict__ vt_sw,
    const float* __restrict__ bias,
    int n_half)
{
  extern __shared__ char smem[];                       // [0,64K): B bufs
  unsigned short* Xb = (unsigned short*)(smem + 2 * BBUF_BYTES);

  const int wave  = threadIdx.x >> 5;
  const int lane  = threadIdx.x & 31;
  const int node0 = (blockIdx.x * WAVES + wave) * M_PER_WAVE;
  unsigned short* Xw = Xb + wave * (M_PER_WAVE * XB_STRIDE);

  // ---- phase 1: X (32 rows x 128) as bf16 into LDS ----
  for (int i = lane; i < M_PER_WAVE * 128; i += 32) {
    int m = i >> 7, c = i & 127;
    int node = node0 + m;
    int b = node / n_half, j = node % n_half;
    int prow = b * (2 * n_half) + 2 * j + (c >> 6);
    float v = nodes_in[(size_t)prow * 64 + (c & 63)];
    unsigned bits = __float_as_uint(v);
    unsigned r = bits + 0x7fffu + ((bits >> 16) & 1u);
    Xw[m * XB_STRIDE + c] = (unsigned short)(r >> 16);
  }
  __syncthreads();

  // ---- phase 2: k-invariant registers ----
  const int m16 = lane & 15;
  const int hi8 = (lane >= 16) ? 8 : 0;

  // xdu[set][dblk][i]: packed bf16 pair of x[m][dblk*32 + koffA(lane,2i..2i+1)]
  // (contiguous mapping k = 16h + e) — also serves verbatim as A-fragments of the W-chunks.
  unsigned xdu[2][4][8];
#pragma unroll
  for (int ws = 0; ws < 2; ++ws) {
    const unsigned short* row = Xw + (ws * 16 + m16) * XB_STRIDE;
#pragma unroll
    for (int dblk = 0; dblk < 4; ++dblk)
#pragma unroll
      for (int i = 0; i < 8; ++i) {
        int koff = 2 * i + ((lane >= 16) ? 16 : 0);
        xdu[ws][dblk][i] =
            *reinterpret_cast<const unsigned*>(row + dblk * 32 + koff);
      }
  }
  float biasr[4];
#pragma unroll
  for (int ni = 0; ni < 4; ++ni) biasr[ni] = bias[ni * 16 + m16];

  // ---- async DMA: stage 8 chunks (32 KB) into buffer ----
  auto fill_async = [&](int dstbuf, int stage) {
    const unsigned short* src = vt_sw + (size_t)stage * STAGE_CH * CHUNK_USH;
    unsigned ldsbase = (unsigned)(size_t)(smem + dstbuf * BBUF_BYTES);
#pragma unroll
    for (int i = 0; i < STAGE_CH * 2; ++i) {     // 2048 x 16B / 128 thr
      int u = (int)threadIdx.x + TPB * i;
      asm volatile("global_load_async_to_lds_b128 %0, %1, off"
                   :: "v"(ldsbase + u * 16),
                      "v"((unsigned long long)(src + u * 8)) : "memory");
    }
  };
  auto wait_async = [] {
#if __has_builtin(__builtin_amdgcn_s_wait_asynccnt)
    __builtin_amdgcn_s_wait_asynccnt(0);
#else
    asm volatile("s_wait_asynccnt 0x0" ::: "memory");
#endif
  };

  union AU { unsigned u[8]; v16bf v; };
  v8f acc[2][4] = {};
  unsigned xcpair[2] = {0, 0};

  fill_async(0, 0);
  wait_async();
  __syncthreads();

  for (int s = 0; s < NSTAGE_FULL; ++s) {
    const int buf = s & 1;
    fill_async(buf ^ 1, s + 1);                  // overlap DMA with compute

    const char* bb = (const char*)smem + buf * BBUF_BYTES + lane * 32;

    v16bf bf[4];                                 // chunk-0 fragments preloaded
#pragma unroll
    for (int ni = 0; ni < 4; ++ni)
      bf[ni] = *reinterpret_cast<const v16bf*>(bb + ni * 1024);

#pragma unroll
    for (int tl = 0; tl < STAGE_CH; ++tl) {
      if ((tl & 3) == 0) {                       // new c every 4 chunks
        int c = s * 2 + (tl >> 2);
#pragma unroll
        for (int ws = 0; ws < 2; ++ws) {
          unsigned short raw = Xw[(ws * 16 + m16) * XB_STRIDE + c];
          xcpair[ws] = (unsigned)raw * 0x10001u; // splat bf16 pair
        }
      }
      const int dblk = tl & 3;
      v16bf Af[2];
#pragma unroll
      for (int ws = 0; ws < 2; ++ws) {           // A = x[c] * xd (packed bf16)
        AU A;
#pragma unroll
        for (int i = 0; i < 8; ++i)
          asm("v_pk_mul_bf16 %0, %1, %2"
              : "=v"(A.u[i]) : "v"(xcpair[ws]), "v"(xdu[ws][dblk][i]));
        Af[ws] = A.v;
      }
      // prefetch next chunk's fragments before consuming this chunk's
      v16bf bfn[4];
      if (tl < STAGE_CH - 1) {
#pragma unroll
        for (int ni = 0; ni < 4; ++ni)
          bfn[ni] = *reinterpret_cast<const v16bf*>(bb + ((tl + 1) * 4 + ni) * 1024);
      }
#pragma unroll
      for (int ni = 0; ni < 4; ++ni) {           // one B frag feeds 2 WMMAs
        acc[0][ni] = __builtin_amdgcn_wmma_f32_16x16x32_bf16(
            false, Af[0], false, bf[ni], (short)0, acc[0][ni], false, false);
        acc[1][ni] = __builtin_amdgcn_wmma_f32_16x16x32_bf16(
            false, Af[1], false, bf[ni], (short)0, acc[1][ni], false, false);
      }
      if (tl < STAGE_CH - 1) {
#pragma unroll
        for (int ni = 0; ni < 4; ++ni) bf[ni] = bfn[ni];
      }
    }
    wait_async();
    __syncthreads();
  }

  // ---- W-stage (chunks 0..3 of stage-64 buffer 0): A = plain x over c ----
  {
    const char* bb = (const char*)smem + lane * 32;
#pragma unroll
    for (int ci = 0; ci < 4; ++ci) {
      v16bf Af[2];
#pragma unroll
      for (int ws = 0; ws < 2; ++ws) {
        AU A;
#pragma unroll
        for (int i = 0; i < 8; ++i) A.u[i] = xdu[ws][ci][i];
        Af[ws] = A.v;
      }
#pragma unroll
      for (int ni = 0; ni < 4; ++ni) {
        v16bf bfrag =
            *reinterpret_cast<const v16bf*>(bb + (ci * 4 + ni) * 1024);
        acc[0][ni] = __builtin_amdgcn_wmma_f32_16x16x32_bf16(
            false, Af[0], false, bfrag, (short)0, acc[0][ni], false, false);
        acc[1][ni] = __builtin_amdgcn_wmma_f32_16x16x32_bf16(
            false, Af[1], false, bfrag, (short)0, acc[1][ni], false, false);
      }
    }
  }

  // ---- epilogue: C layout == bil[m,k]; tanh(acc + bias) -> global ----
#pragma unroll
  for (int ws = 0; ws < 2; ++ws)
#pragma unroll
    for (int ni = 0; ni < 4; ++ni)
#pragma unroll
      for (int r = 0; r < 8; ++r) {
        float v = tanhf(acc[ws][ni][r] + biasr[ni]);
        int node = node0 + ws * 16 + hi8 + r;
        nodes_out[(size_t)node * 64 + ni * 16 + m16] = v;
      }
}

// ---------------------------------------------------------------------------
__global__ __launch_bounds__(256) void out_kernel(
    const float* __restrict__ nodes, const float* __restrict__ W_out,
    float* __restrict__ out)
{
  __shared__ float Ws[64 * 5];
  for (int i = threadIdx.x; i < 64 * 5; i += 256) Ws[i] = W_out[i];
  __syncthreads();

  int row = blockIdx.x * 256 + threadIdx.x;    // exact: 130048 = 508*256
  const float* x = nodes + (size_t)row * 64;
  float logit[5] = {0.f, 0.f, 0.f, 0.f, 0.f};
  for (int c = 0; c < 64; ++c) {
    float xv = x[c];
#pragma unroll
    for (int o = 0; o < 5; ++o) logit[o] += xv * Ws[c * 5 + o];
  }
  float mx = logit[0];
#pragma unroll
  for (int o = 1; o < 5; ++o) mx = fmaxf(mx, logit[o]);
  float se = 0.0f;
#pragma unroll
  for (int o = 0; o < 5; ++o) se += expf(logit[o] - mx);
  float lse = logf(se) + mx;
#pragma unroll
  for (int o = 0; o < 5; ++o) out[(size_t)row * 5 + o] = logit[o] - lse;
}

// ---------------------------------------------------------------------------
extern "C" void kernel_launch(void* const* d_in, const int* in_sizes, int n_in,
                              void* d_out, int out_size, void* d_ws, size_t ws_size,
                              hipStream_t stream)
{
  const int*   leaf_ids = (const int*)  d_in[0];
  const float* embed    = (const float*)d_in[1];
  const float* Vt       = (const float*)d_in[2];
  const float* W        = (const float*)d_in[3];
  const float* bvec     = (const float*)d_in[4];
  const float* W_out    = (const float*)d_in[5];
  float* out = (float*)d_out;

  // workspace: nodes f32 (33.3 MB) | B-stream bf16 (2.1 MB)
  float* nodes = (float*)d_ws;
  unsigned short* vt_sw =
      (unsigned short*)((char*)d_ws + (size_t)NODES_TOTAL * 64 * sizeof(float));

  vt_swizzle_kernel<<<VT2_ELEMS / 256, 256, 0, stream>>>(Vt, W, vt_sw);
  gather_kernel<<<(LEAF_ROWS * 64) / 256, 256, 0, stream>>>(leaf_ids, embed, nodes);

  static const int level_off[8] = {0, 65536, 98304, 114688, 122880, 126976, 129024, 130048};
  for (int lvl = 0; lvl < 6; ++lvl) {
    int n_half = 32 >> lvl;
    const float* nin = nodes + (size_t)level_off[lvl] * 64;
    float* nout      = nodes + (size_t)level_off[lvl + 1] * 64;
    int blocks = (B_DIM * n_half) / NODES_PER_BLK;
    level_kernel<<<blocks, TPB, SMEM_LVL, stream>>>(nin, nout, vt_sw, bvec, n_half);
  }

  out_kernel<<<NODES_TOTAL / 256, 256, 0, stream>>>(nodes, W_out, out);
}
